// GNN_classifier_79826262164185
// MI455X (gfx1250) — matrix operand order repaired and checked by reading, not compiled, not run on hardware
//
#include <hip/hip_runtime.h>
#include <hip/hip_bf16.h>

#define NN 100000   // nodes
#define EE 800000   // edges
#define DD 128      // feature dim
#define LL 4        // layers
#define GG 64       // graphs
#define LABELS 4
#define ROWS 80     // M rows per GEMM block (100000 = 1250 * 80)
#define MT   5      // 16-row sub-tiles per wave (ROWS / 16)

typedef __bf16 bf16;
typedef __attribute__((ext_vector_type(16))) __bf16 v16bf;
typedef __attribute__((ext_vector_type(8)))  float  v8f;

// ---------------- degree / normalization ----------------
__global__ void __launch_bounds__(256) k_init_deg(float* deg) {
    int i = blockIdx.x * 256 + threadIdx.x;
    if (i < NN) deg[i] = 1.0f;                    // self loop
}
__global__ void __launch_bounds__(256) k_count_deg(const int* __restrict__ dst, float* deg) {
    int e = blockIdx.x * 256 + threadIdx.x;       // EE % 256 == 0
    atomicAdd(&deg[dst[e]], 1.0f);
}
__global__ void __launch_bounds__(256) k_finish_dis(float* deg) {
    int i = blockIdx.x * 256 + threadIdx.x;
    if (i < NN) deg[i] = rsqrtf(fmaxf(deg[i], 1.0f));
}

// ---------------- f32 -> bf16 converter ----------------
__global__ void __launch_bounds__(256) k_f2bf(const float* __restrict__ src, bf16* __restrict__ dst) {
    int i = blockIdx.x * 256 + threadIdx.x;       // callers size grid exactly
    dst[i] = (bf16)src[i];
}

// ---------------- WMMA GEMM: t = h @ W[i]; agg = t*dis^2 + b[i] ----------------
// 80 M-rows per block, 8 waves each owning a 16-wide N strip; per wave:
// 5 independent accumulator chains x 4 K-steps = 20 v_wmma_f32_16x16x32_bf16.
__global__ void __launch_bounds__(256)
k_gemm(const bf16* __restrict__ hbf, const bf16* __restrict__ Wl,
       const float* __restrict__ bias, const float* __restrict__ dis,
       float* __restrict__ t, float* __restrict__ agg)
{
    __shared__ bf16 sW[DD * DD];      // 32 KB, row-major W[k][n]
    __shared__ bf16 sA[ROWS * DD];    // 20 KB, 80 rows of h (bf16)
    const int tid   = threadIdx.x;
    const int tileM = blockIdx.x * ROWS;

    // stage W (2048 uint4) and A tile (1280 uint4) into LDS
    {
        const uint4* wsrc = (const uint4*)Wl;
        uint4*       wdst = (uint4*)sW;
        #pragma unroll
        for (int i = 0; i < 8; ++i) wdst[tid + 256 * i] = wsrc[tid + 256 * i];
        const uint4* asrc = (const uint4*)(hbf + (size_t)tileM * DD);
        uint4*       adst = (uint4*)sA;
        #pragma unroll
        for (int i = 0; i < MT; ++i) adst[tid + 256 * i] = asrc[tid + 256 * i];
    }
    __syncthreads();

    const int wave  = tid >> 5;       // 0..7 -> N strip
    const int lane  = tid & 31;
    const int tileN = wave * 16;
    const int m16   = lane & 15;
    const int half  = lane >> 4;

    v8f c[MT] = {};                   // 5 independent f32 accumulator chains
    #pragma unroll
    for (int s = 0; s < 4; ++s) {
        const int kb = s * 32;
        // B frag (shared across the 5 M sub-tiles): lane n = tileN+m16; elems 0..7 -> K = kb+half*8+0..7; elems 8..15 -> +16
        v16bf bm;
        const bf16* bcol = sW + (kb + half * 8) * DD + tileN + m16;
        #pragma unroll
        for (int j = 0; j < 8; ++j) { bm[j] = bcol[j * DD]; bm[j + 8] = bcol[(j + 16) * DD]; }
        #pragma unroll
        for (int mt = 0; mt < MT; ++mt) {
            // A frag: lane m = m16; elems 0..7 -> K = kb+half*8+0..7; elems 8..15 -> +16
            v16bf a;
            const bf16* arow = sA + (mt * 16 + m16) * DD + kb + half * 8;
            *((uint4*)&a)       = *(const uint4*)(arow);
            *(((uint4*)&a) + 1) = *(const uint4*)(arow + 16);
            c[mt] = __builtin_amdgcn_wmma_f32_16x16x32_bf16(false, a, false, bm,
                                                            (short)0, c[mt], false, false);
        }
    }

    // C/D layout: lane -> n = lane%16 ; VGPR r -> m = r + 8*(lane/16)
    const int   n  = tileN + m16;
    const float bn = bias[n];
    #pragma unroll
    for (int mt = 0; mt < MT; ++mt) {
        #pragma unroll
        for (int r = 0; r < 8; ++r) {
            int    m  = tileM + mt * 16 + r + 8 * half;
            float  v  = c[mt][r];
            float  ds = dis[m];
            t[(size_t)m * DD + n]   = v;
            agg[(size_t)m * DD + n] = v * ds * ds + bn;   // self-loop + bias
        }
    }
}

// ---------------- edge scatter: agg[dst] += t[src] * norm ----------------
__global__ void __launch_bounds__(256)
k_scatter(const int* __restrict__ src, const int* __restrict__ dst,
          const float* __restrict__ dis, const float* __restrict__ t,
          float* __restrict__ agg)
{
    int wid  = (blockIdx.x * 256 + threadIdx.x) >> 5;   // one wave per edge
    int lane = threadIdx.x & 31;
    if (wid >= EE) return;
    int   s = src[wid], d = dst[wid];
    float w = dis[s] * dis[d];
    float4 v = ((const float4*)(t + (size_t)s * DD))[lane];
    float* ag = agg + (size_t)d * DD + lane * 4;
    atomicAdd(ag + 0, v.x * w);
    atomicAdd(ag + 1, v.y * w);
    atomicAdd(ag + 2, v.z * w);
    atomicAdd(ag + 3, v.w * w);
}

// ---------------- residual + ReLU (+ global stats for LayerNorm) ----------------
__global__ void __launch_bounds__(256)
k_post1(const float* __restrict__ agg, const float* __restrict__ prev,
        float* __restrict__ h, float* __restrict__ stats, int do_stats)
{
    int base = blockIdx.x * 2048 + threadIdx.x;   // 6250 * 2048 == NN*DD
    float s = 0.f, s2 = 0.f;
    #pragma unroll
    for (int i = 0; i < 8; ++i) {
        int   idx = base + i * 256;
        float v   = agg[idx] + prev[idx];
        v = v > 0.f ? v : 0.f;
        h[idx] = v;
        s += v; s2 += v * v;
    }
    if (do_stats) {
        for (int off = 16; off; off >>= 1) {
            s  += __shfl_down(s,  off, 32);
            s2 += __shfl_down(s2, off, 32);
        }
        __shared__ float red[16];
        int lane = threadIdx.x & 31, wv = threadIdx.x >> 5;
        if (lane == 0) { red[wv] = s; red[wv + 8] = s2; }
        __syncthreads();
        if (threadIdx.x == 0) {
            float ts = 0.f, ts2 = 0.f;
            for (int i = 0; i < 8; ++i) { ts += red[i]; ts2 += red[i + 8]; }
            atomicAdd(&stats[0], ts);
            atomicAdd(&stats[1], ts2);
        }
    }
}

// ---------------- graph-LayerNorm apply; also emit bf16 for next GEMM ----------------
__global__ void __launch_bounds__(256)
k_post2(float* __restrict__ h, const float* __restrict__ stats,
        const float* __restrict__ lw, const float* __restrict__ lb,
        bf16* __restrict__ hbf)
{
    const float inv  = 1.0f / ((float)NN * (float)DD);
    float mu   = stats[0] * inv;
    float var  = stats[1] * inv - mu * mu;
    float rstd = rsqrtf(var + 1e-5f);
    int base = blockIdx.x * 2048 + threadIdx.x;
    #pragma unroll
    for (int i = 0; i < 8; ++i) {
        int   idx = base + i * 256;
        int   d   = idx & (DD - 1);
        float v   = (h[idx] - mu) * rstd * lw[d] + lb[d];
        h[idx]   = v;
        hbf[idx] = (bf16)v;
    }
}

// ---------------- global mean pool (atomics) ----------------
__global__ void __launch_bounds__(256)
k_pool(const float* __restrict__ h, const int* __restrict__ batch,
       float* __restrict__ pooled, float* __restrict__ counts)
{
    int node = (blockIdx.x * 256 + threadIdx.x) >> 5;   // 12500 blocks * 8 waves == NN
    int lane = threadIdx.x & 31;
    int g    = batch[node];
    float4 v = ((const float4*)(h + (size_t)node * DD))[lane];
    float* p = pooled + g * DD + lane * 4;
    atomicAdd(p + 0, v.x);
    atomicAdd(p + 1, v.y);
    atomicAdd(p + 2, v.z);
    atomicAdd(p + 3, v.w);
    if (lane == 0) atomicAdd(&counts[g], 1.0f);
}

// ---------------- MLP head (tiny, single block) ----------------
__global__ void __launch_bounds__(256)
k_mlp(const float* __restrict__ pooled, const float* __restrict__ counts,
      const float* __restrict__ w1, const float* __restrict__ b1,
      const float* __restrict__ w2, const float* __restrict__ b2,
      float* __restrict__ out)
{
    __shared__ float sp[GG * DD];   // 32 KB
    int tid = threadIdx.x;
    #pragma unroll
    for (int i = 0; i < 32; ++i) {             // 8192 = 256*32
        int idx = tid + i * 256;
        int g   = idx >> 7;
        sp[idx] = pooled[idx] / fmaxf(counts[g], 1.0f);
    }
    __syncthreads();
    float hid[32];
    #pragma unroll
    for (int i = 0; i < 32; ++i) {
        int idx = tid + i * 256;
        int g = idx >> 7, j = idx & 127;
        float acc = b1[j];
        for (int k = 0; k < DD; ++k) acc += sp[g * DD + k] * w1[k * DD + j];
        hid[i] = fmaxf(acc, 0.0f);
    }
    __syncthreads();
    #pragma unroll
    for (int i = 0; i < 32; ++i) sp[tid + i * 256] = hid[i];
    __syncthreads();
    {
        int g = tid >> 2, c = tid & 3;
        float acc = b2[c];
        for (int j = 0; j < DD; ++j) acc += sp[g * DD + j] * w2[j * LABELS + c];
        out[g * LABELS + c] = acc;
    }
}

// ---------------- host launch ----------------
extern "C" void kernel_launch(void* const* d_in, const int* in_sizes, int n_in,
                              void* d_out, int out_size, void* d_ws, size_t ws_size,
                              hipStream_t stream)
{
    const float* x     = (const float*)d_in[0];   // [N,D]
    const int*   edge  = (const int*)  d_in[1];   // [2,E]
    const int*   batch = (const int*)  d_in[2];   // [N]
    const float* W     = (const float*)d_in[3];   // [L,D,D]
    const float* b     = (const float*)d_in[4];   // [L,D]
    const float* ln_w  = (const float*)d_in[5];   // [L-1,D]
    const float* ln_b  = (const float*)d_in[6];
    const float* w1    = (const float*)d_in[7];
    const float* b1    = (const float*)d_in[8];
    const float* w2    = (const float*)d_in[9];
    const float* b2    = (const float*)d_in[10];
    float*       out   = (float*)d_out;

    const int* e_src = edge;
    const int* e_dst = edge + EE;

    // workspace carve-up
    char*  ws  = (char*)d_ws;
    size_t off = 0;
    auto alloc = [&](size_t bytes) -> char* {
        char* p = ws + off;
        off += (bytes + 255) & ~(size_t)255;
        return p;
    };
    float* dis    = (float*)alloc((size_t)NN * 4);
    float* t      = (float*)alloc((size_t)NN * DD * 4);
    float* agg    = (float*)alloc((size_t)NN * DD * 4);
    float* h      = (float*)alloc((size_t)NN * DD * 4);
    bf16*  hbf    = (bf16*) alloc((size_t)NN * DD * 2);
    bf16*  Wbf    = (bf16*) alloc((size_t)LL * DD * DD * 2);
    float* stats  = (float*)alloc(2 * 4);
    float* pooled = (float*)alloc((size_t)GG * DD * 4);
    float* counts = (float*)alloc((size_t)GG * 4);

    // normalization: dis = rsqrt(deg), deg includes self loop
    k_init_deg  <<<(NN + 255) / 256, 256, 0, stream>>>(dis);
    k_count_deg <<<EE / 256,         256, 0, stream>>>(e_dst, dis);
    k_finish_dis<<<(NN + 255) / 256, 256, 0, stream>>>(dis);

    // bf16 copies of weights (all layers) and initial features
    k_f2bf<<<(LL * DD * DD) / 256, 256, 0, stream>>>(W, Wbf);
    k_f2bf<<<(NN * DD) / 256,      256, 0, stream>>>(x, hbf);

    for (int i = 0; i < LL; ++i) {
        const float* prev = (i == 0) ? x : h;
        hipMemsetAsync(stats, 0, 2 * sizeof(float), stream);

        k_gemm   <<<NN / ROWS, 256, 0, stream>>>(hbf, Wbf + (size_t)i * DD * DD,
                                                 b + i * DD, dis, t, agg);
        k_scatter<<<EE / 8,    256, 0, stream>>>(e_src, e_dst, dis, t, agg);
        k_post1  <<<(NN * DD) / 2048, 256, 0, stream>>>(agg, prev, h, stats,
                                                        (i < LL - 1) ? 1 : 0);
        if (i < LL - 1)
            k_post2<<<(NN * DD) / 2048, 256, 0, stream>>>(h, stats,
                                                          ln_w + i * DD, ln_b + i * DD, hbf);
    }

    // pooling + head
    hipMemsetAsync(pooled, 0, (size_t)GG * DD * sizeof(float), stream);
    hipMemsetAsync(counts, 0, (size_t)GG * sizeof(float), stream);
    k_pool<<<NN / 8, 256, 0, stream>>>(h, batch, pooled, counts);
    k_mlp <<<1,      256, 0, stream>>>(pooled, counts, w1, b1, w2, b2, out);
}
